// SimpleTSSGCNet_9620726743376
// MI455X (gfx1250) — matrix operand order repaired and checked by reading, not compiled, not run on hardware
//
#include <hip/hip_runtime.h>
#include <hip/hip_bf16.h>

// ---------------------------------------------------------------------------
// SimpleTSSGCNet forward on MI455X (gfx1250), bf16 WMMA + f32 accumulate.
// N=50000, K=16, D=128, H=64, 3H=192, C=2.
// dst = repeat(arange(N),K)  =>  deg == 17 everywhere, sym-norm == 1/17.
// ---------------------------------------------------------------------------

#define NN   50000
#define KNB  16
#define DD   128
#define HH   64
#define GG   192            // 3*H
#define CC   2
#define INV17 (1.0f / 17.0f)
#define BN_EPS 1e-5f

typedef __attribute__((ext_vector_type(16))) __bf16 v16bf;
typedef __attribute__((ext_vector_type(4)))  __bf16 v4bf;
typedef __attribute__((ext_vector_type(2)))  __bf16 v2bf;
typedef __attribute__((ext_vector_type(8)))  float  v8f;

__device__ __forceinline__ v8f wmma_bf16(v16bf a, v16bf b, v8f c) {
  // D = A(16x32 bf16) x B(32x16 bf16) + C(16x16 f32)
  return __builtin_amdgcn_wmma_f32_16x16x32_bf16(
      /*neg_a=*/false, a, /*neg_b=*/false, b,
      /*c_mod=*/(short)0, c, /*reuse_a=*/false, /*reuse_b=*/false);
}

// A-fragment element -> K index (this code's contiguous order, shared by A and B):
// lane = half*16 + m; element j; K = 16*half + j
__device__ __forceinline__ int a_k_of(int j, int half) {
  return 16 * half + j;
}
// Inverse: in-tile K index kk (0..31) -> (half, j).  Contiguous for kk%16<16 runs.
__device__ __forceinline__ void a_inv_of(int kk, int& half, int& j) {
  half = kk >> 4;
  j = kk & 15;
}

__device__ __forceinline__ __bf16 f2bf(float f) { return (__bf16)f; }

// v_rcp_f32-based fast activations (feed saturating GRU gates; plenty accurate)
__device__ __forceinline__ float fast_sigmoid(float x) {
  return __builtin_amdgcn_rcpf(1.0f + __expf(-x));
}
__device__ __forceinline__ float fast_tanh(float x) {
  return 1.0f - 2.0f * __builtin_amdgcn_rcpf(__expf(2.0f * x) + 1.0f);
}

// gfx1250 async global->LDS copy (16B per lane), tracked by ASYNCcnt.
__device__ __forceinline__ void async_copy_b128(const __bf16* gsrc, __bf16* lds_dst) {
  unsigned lds = (unsigned)(size_t)lds_dst;                   // LDS offset = addr[31:0]
  unsigned long long src = (unsigned long long)(size_t)gsrc;
  asm volatile("global_load_async_to_lds_b128 %0, %1, off"
               :: "v"(lds), "v"(src) : "memory");
}
__device__ __forceinline__ void wait_async0() {
  asm volatile("s_wait_asynccnt 0x0" ::: "memory");
}

// ---------------------------------------------------------------------------
// Conversions / init.  Weights for k_gemm are stored TRANSPOSED (Bt[n][k]) so
// each lane's WMMA B-fragment is one contiguous 32B load.
// ---------------------------------------------------------------------------
__global__ void k_cvt_x(const float* __restrict__ x, __bf16* __restrict__ xb, int n) {
  for (int i = blockIdx.x * blockDim.x + threadIdx.x; i < n;
       i += gridDim.x * blockDim.x)
    xb[i] = f2bf(x[i]);
}

__global__ void k_cvt_w(const float* __restrict__ W1, const float* __restrict__ W2,
                        const float* __restrict__ Wih, const float* __restrict__ Whh,
                        __bf16* __restrict__ W1t, __bf16* __restrict__ W2t,
                        __bf16* __restrict__ Wihb, __bf16* __restrict__ Whhb,
                        float* __restrict__ stats) {
  int tid = blockIdx.x * blockDim.x + threadIdx.x;
  int stride = gridDim.x * blockDim.x;
  for (int i = tid; i < DD * HH; i += stride) {   // W1t[n][k] = W1[k][n]
    int n = i / DD, k = i % DD;
    W1t[i] = f2bf(W1[(size_t)k * HH + n]);
  }
  for (int i = tid; i < HH * HH; i += stride) {   // W2t[n][k] = W2[k][n]
    int n = i / HH, k = i % HH;
    W2t[i] = f2bf(W2[(size_t)k * HH + n]);
  }
  for (int i = tid; i < GG * DD; i += stride) Wihb[i] = f2bf(Wih[i]); // already [g][d]
  for (int i = tid; i < GG * HH; i += stride) Whhb[i] = f2bf(Whh[i]); // already [g][h]
  for (int i = tid; i < 4 * HH; i += stride)  stats[i] = 0.0f;
}

// ---------------------------------------------------------------------------
// C[M,64] = A[M,KD](bf16) @ Bt^T (Bt is [64][KD] bf16), f32 acc via WMMA.
// Block: 128 threads (4 waves); 16-row M tile per block, wave w owns N-tile w.
// A tile staged via async global->LDS copy; A-fragments read as 8x b32 pairs.
// ---------------------------------------------------------------------------
template <int KD>
__global__ __launch_bounds__(128)
void k_gemm(const __bf16* __restrict__ A, const __bf16* __restrict__ Bt,
            float* __restrict__ C) {
  __shared__ __attribute__((aligned(32))) __bf16 Alds[16 * KD];
  const int tid  = threadIdx.x;
  const int row0 = blockIdx.x * 16;

  // async copy 16B chunks: 16*KD*2 bytes total
  constexpr int NCHUNK = (16 * KD * 2) / 16;   // == 2*KD
#pragma unroll
  for (int c0 = 0; c0 < NCHUNK; c0 += 128) {
    int c = c0 + tid;
    if (c < NCHUNK)
      async_copy_b128(A + (size_t)row0 * KD + c * 8, &Alds[c * 8]);
  }
  wait_async0();
  __syncthreads();

  const int lane = tid & 31;
  const int half = lane >> 4;       // 0 | 1
  const int col  = lane & 15;       // N column inside tile / A row
  const int nt   = (tid >> 5) * 16; // this wave's N-tile base (0,16,32,48)

  v8f acc = {};
#pragma unroll
  for (int kb = 0; kb < KD; kb += 32) {
    union { v16bf v; v2bf p[8]; } af;
#pragma unroll
    for (int i = 0; i < 8; ++i)     // pairs a_k_of(2i),a_k_of(2i+1) are contiguous
      af.p[i] = *(const v2bf*)&Alds[col * KD + kb + a_k_of(2 * i, half)];
    // B[d][n] = Bt[n][d]: contiguous 16 bf16 (32B) per lane
    v16bf b = *(const v16bf*)&Bt[(size_t)(nt + col) * KD + kb + 16 * half];
    acc = wmma_bf16(af.v, b, acc);
  }
#pragma unroll
  for (int j = 0; j < 8; ++j) {
    int m = row0 + j + 8 * half;
    C[(size_t)m * HH + nt + col] = acc[j];
  }
}

// ---------------------------------------------------------------------------
// GCN aggregate: out = (sum_{16 nbrs} h[src] + h[self]) / 17 + bias,
// plus block-partial BN statistics -> global atomics.
// One wave per node; lane handles features {2*lane, 2*lane+1}.
// ---------------------------------------------------------------------------
__global__ __launch_bounds__(256)
void k_aggregate(const float* __restrict__ h, const int* __restrict__ edge_src,
                 const float* __restrict__ bias, float* __restrict__ out,
                 float* __restrict__ stats) {
  __shared__ float s_sum[HH];
  __shared__ float s_sq[HH];
  const int tid = threadIdx.x;
  if (tid < HH) { s_sum[tid] = 0.0f; s_sq[tid] = 0.0f; }
  __syncthreads();

  const int node = blockIdx.x * 8 + (tid >> 5);
  const int lane = tid & 31;
  const int f0   = lane * 2;

  const float2 self = *(const float2*)&h[(size_t)node * HH + f0];
  float ax = self.x, ay = self.y;
  const int* nb = &edge_src[(size_t)node * KNB];
#pragma unroll
  for (int k = 0; k < KNB; ++k) {
    int s = nb[k];
    if (k + 1 < KNB) __builtin_prefetch(&h[(size_t)nb[k + 1] * HH + f0], 0, 0);
    float2 v = *(const float2*)&h[(size_t)s * HH + f0];
    ax += v.x; ay += v.y;
  }
  float rx = ax * INV17 + bias[f0];
  float ry = ay * INV17 + bias[f0 + 1];
  float2 r; r.x = rx; r.y = ry;
  *(float2*)&out[(size_t)node * HH + f0] = r;

  atomicAdd(&s_sum[f0], rx);       atomicAdd(&s_sum[f0 + 1], ry);
  atomicAdd(&s_sq[f0],  rx * rx);  atomicAdd(&s_sq[f0 + 1],  ry * ry);
  __syncthreads();
  if (tid < HH) {
    atomicAdd(&stats[tid],      s_sum[tid]);
    atomicAdd(&stats[HH + tid], s_sq[tid]);
  }
}

// ---------------------------------------------------------------------------
// BatchNorm (batch stats, biased var) + ReLU; writes f32 and/or bf16.
// ---------------------------------------------------------------------------
__global__ void k_bn_relu(const float* __restrict__ v, const float* __restrict__ stats,
                          const float* __restrict__ g, const float* __restrict__ be,
                          float* __restrict__ out_f32, __bf16* __restrict__ out_bf16) {
  int idx = blockIdx.x * blockDim.x + threadIdx.x;
  if (idx >= NN * HH) return;
  int f = idx & (HH - 1);
  float mu  = stats[f] * (1.0f / NN);
  float var = stats[HH + f] * (1.0f / NN) - mu * mu;
  float y = (v[idx] - mu) * rsqrtf(var + BN_EPS) * g[f] + be[f];
  y = fmaxf(y, 0.0f);
  if (out_f32)  out_f32[idx]  = y;
  if (out_bf16) out_bf16[idx] = f2bf(y);
}

// ---------------------------------------------------------------------------
// Stage one GRU step's alpha-weighted neighbor features into a fragment-order
// LDS buffer.  float4 global loads -> packed 4x bf16 LDS stores (layout keeps
// 4-aligned K runs contiguous).
// ---------------------------------------------------------------------------
__device__ __forceinline__ void stage_feats(const float* __restrict__ x,
                                            const int* s_nb, const float* s_alpha,
                                            __bf16* buf, int kstep, int tid) {
  for (int idx = tid; idx < 16 * (DD / 4); idx += 384) {   // 512 chunks of 4
    int n = idx >> 5, d = (idx & 31) * 4;
    int nb = s_nb[n * KNB + kstep];
    float al = s_alpha[n * KNB + kstep];
    float4 v = *(const float4*)&x[(size_t)nb * DD + d];
    int ks = d >> 5, kk = d & 31, hf, j;
    a_inv_of(kk, hf, j);
    v4bf o;
    o[0] = f2bf(v.x * al); o[1] = f2bf(v.y * al);
    o[2] = f2bf(v.z * al); o[3] = f2bf(v.w * al);
    *(v4bf*)&buf[ks * 512 + (hf * 16 + n) * 16 + j] = o;
  }
}

// ---------------------------------------------------------------------------
// Fused temporal branch: time-decay attention + 16-step GRU.
// Block: 384 threads = 12 waves; block owns 16 nodes; wave w owns gate tile
// [w*16, w*16+16).  A operands staged in WMMA-fragment order (32B/lane reads);
// feats tile double-buffered so the next step's gather overlaps this step's
// matrix + gate work.  2 barriers per step.
// ---------------------------------------------------------------------------
__global__ __launch_bounds__(384)
void k_temporal(const float* __restrict__ x, const float* __restrict__ ts,
                const int* __restrict__ edge_src,
                const __bf16* __restrict__ Wihb, const __bf16* __restrict__ Whhb,
                const float* __restrict__ bih, const float* __restrict__ bhh,
                const float* __restrict__ beta_p, float* __restrict__ hT) {
  __shared__ int    s_nb[16 * KNB];
  __shared__ float  s_alpha[16 * KNB];
  __shared__ __attribute__((aligned(32))) __bf16 s_featf[2][4 * 512]; // dbl buf
  __shared__ __attribute__((aligned(32))) __bf16 s_hbf[2 * 512];      // K=64
  __shared__ float  s_h[16][HH];
  __shared__ float  s_gi[16][GG];
  __shared__ float  s_gh[16][GG];

  const int tid   = threadIdx.x;
  const int node0 = blockIdx.x * 16;
  const float beta = beta_p[0];

  // neighbors + unnormalized attention
  if (tid < 16 * KNB) {
    int n = tid >> 4, k = tid & 15;
    int nb = edge_src[(size_t)(node0 + n) * KNB + k];
    s_nb[n * KNB + k] = nb;
    float dt = fmaxf(ts[node0 + n] - ts[nb], 0.0f);
    s_alpha[n * KNB + k] = __expf(-beta * dt);
  }
  for (int idx = tid; idx < 16 * HH; idx += 384) {
    int n = idx >> 6, hh = idx & 63;
    s_h[n][hh] = 0.0f;
  }
  for (int idx = tid; idx < 2 * 512; idx += 384) s_hbf[idx] = f2bf(0.0f);
  __syncthreads();
  if (tid < 16) { // normalize alpha per node
    float s = 1e-9f;
#pragma unroll
    for (int k = 0; k < KNB; ++k) s += s_alpha[tid * KNB + k];
    float inv = __builtin_amdgcn_rcpf(s);
#pragma unroll
    for (int k = 0; k < KNB; ++k) s_alpha[tid * KNB + k] *= inv;
  }
  __syncthreads();

  stage_feats(x, s_nb, s_alpha, s_featf[0], 0, tid);
  __syncthreads();

  const int lane  = tid & 31;
  const int half  = lane >> 4;
  const int col   = lane & 15;
  const int gbase = (tid >> 5) * 16;          // wave's gate tile base, 0..176
  const float bi = bih[gbase + col];
  const float bh = bhh[gbase + col];

  for (int kstep = 0; kstep < KNB; ++kstep) {
    const __bf16* cur = s_featf[kstep & 1];

    // gi tile: M=16 nodes, N=16 gates, K=128
    v8f gi = { bi, bi, bi, bi, bi, bi, bi, bi };
#pragma unroll
    for (int kb = 0; kb < DD; kb += 32) {
      v16bf a = *(const v16bf*)&cur[(kb >> 5) * 512 + lane * 16];
      v16bf b = *(const v16bf*)&Wihb[(size_t)(gbase + col) * DD + kb + 16 * half];
      gi = wmma_bf16(a, b, gi);
    }
    // gh tile: M=16 nodes, N=16 gates, K=64
    v8f gh = { bh, bh, bh, bh, bh, bh, bh, bh };
#pragma unroll
    for (int kb = 0; kb < HH; kb += 32) {
      v16bf a = *(const v16bf*)&s_hbf[(kb >> 5) * 512 + lane * 16];
      v16bf b = *(const v16bf*)&Whhb[(size_t)(gbase + col) * HH + kb + 16 * half];
      gh = wmma_bf16(a, b, gh);
    }

    // overlap: gather next step's feats into the other buffer
    if (kstep + 1 < KNB)
      stage_feats(x, s_nb, s_alpha, s_featf[(kstep + 1) & 1], kstep + 1, tid);

#pragma unroll
    for (int j = 0; j < 8; ++j) {
      int m = j + 8 * half;
      s_gi[m][gbase + col] = gi[j];
      s_gh[m][gbase + col] = gh[j];
    }
    __syncthreads();

    // GRU gate math (f32), update h (f32 + fragment-order bf16 copy)
    for (int idx = tid; idx < 16 * HH; idx += 384) {
      int n = idx >> 6, hh = idx & 63;
      float r  = fast_sigmoid(s_gi[n][hh] + s_gh[n][hh]);
      float z  = fast_sigmoid(s_gi[n][HH + hh] + s_gh[n][HH + hh]);
      float nn = fast_tanh(s_gi[n][2 * HH + hh] + r * s_gh[n][2 * HH + hh]);
      float hp = s_h[n][hh];
      float hn = (1.0f - z) * nn + z * hp;
      s_h[n][hh] = hn;
      int ks = hh >> 5, kk = hh & 31, hf, j;
      a_inv_of(kk, hf, j);
      s_hbf[ks * 512 + (hf * 16 + n) * 16 + j] = f2bf(hn);
    }
    __syncthreads();
  }

  for (int idx = tid; idx < 16 * HH; idx += 384) {
    int n = idx >> 6, hh = idx & 63;
    hT[(size_t)(node0 + n) * HH + hh] = s_h[n][hh];
  }
}

// ---------------------------------------------------------------------------
// Head: out = concat(h2, hT) @ Wo + bo   ([N,128] @ [128,2])
// ---------------------------------------------------------------------------
__global__ void k_head(const float* __restrict__ h2, const float* __restrict__ hT,
                       const float* __restrict__ Wo, const float* __restrict__ bo,
                       float* __restrict__ out) {
  int n = blockIdx.x * blockDim.x + threadIdx.x;
  if (n >= NN) return;
  float a0 = bo[0], a1 = bo[1];
#pragma unroll 8
  for (int j = 0; j < HH; ++j) {
    float v = h2[(size_t)n * HH + j];
    a0 += v * Wo[j * CC + 0];
    a1 += v * Wo[j * CC + 1];
  }
#pragma unroll 8
  for (int j = 0; j < HH; ++j) {
    float v = hT[(size_t)n * HH + j];
    a0 += v * Wo[(HH + j) * CC + 0];
    a1 += v * Wo[(HH + j) * CC + 1];
  }
  out[(size_t)n * CC + 0] = a0;
  out[(size_t)n * CC + 1] = a1;
}

// ---------------------------------------------------------------------------
extern "C" void kernel_launch(void* const* d_in, const int* in_sizes, int n_in,
                              void* d_out, int out_size, void* d_ws, size_t ws_size,
                              hipStream_t stream) {
  const float* x    = (const float*)d_in[0];
  const float* ts   = (const float*)d_in[1];
  const int*   esrc = (const int*)d_in[2];
  const float* W1   = (const float*)d_in[3];
  const float* b1   = (const float*)d_in[4];
  const float* g1   = (const float*)d_in[5];
  const float* be1  = (const float*)d_in[6];
  const float* W2   = (const float*)d_in[7];
  const float* b2   = (const float*)d_in[8];
  const float* g2   = (const float*)d_in[9];
  const float* be2  = (const float*)d_in[10];
  const float* Wih  = (const float*)d_in[11];
  const float* Whh  = (const float*)d_in[12];
  const float* bih  = (const float*)d_in[13];
  const float* bhh  = (const float*)d_in[14];
  const float* beta = (const float*)d_in[15];
  const float* Wo   = (const float*)d_in[16];
  const float* bo   = (const float*)d_in[17];
  float* out = (float*)d_out;

  // workspace carve-out (256B aligned)
  char* ws = (char*)d_ws;
  size_t off = 0;
  auto carve = [&](size_t bytes) -> char* {
    char* p = ws + off;
    off = (off + bytes + 255) & ~(size_t)255;
    return p;
  };
  __bf16* xb    = (__bf16*)carve((size_t)NN * DD * 2);
  __bf16* W1t   = (__bf16*)carve(HH * DD * 2);   // transposed [64][128]
  __bf16* W2t   = (__bf16*)carve(HH * HH * 2);   // transposed [64][64]
  __bf16* Wihb  = (__bf16*)carve(GG * DD * 2);
  __bf16* Whhb  = (__bf16*)carve(GG * HH * 2);
  float*  stats = (float*)carve(4 * HH * 4);
  float*  hpre  = (float*)carve((size_t)NN * HH * 4);
  float*  agg   = (float*)carve((size_t)NN * HH * 4);
  __bf16* h1b   = (__bf16*)carve((size_t)NN * HH * 2);
  float*  h2    = (float*)carve((size_t)NN * HH * 4);
  float*  hT    = (float*)carve((size_t)NN * HH * 4);
  if (off > ws_size) return; // workspace too small; nothing safe to do

  // ---- prep: bf16 conversions (+ weight transpose) + zero BN stats ----
  k_cvt_x<<<1024, 256, 0, stream>>>(x, xb, NN * DD);
  k_cvt_w<<<64, 256, 0, stream>>>(W1, W2, Wih, Whh, W1t, W2t, Wihb, Whhb, stats);

  // ---- spatial layer 1 ----
  k_gemm<DD><<<NN / 16, 128, 0, stream>>>(xb, W1t, hpre);
  k_aggregate<<<NN / 8, 256, 0, stream>>>(hpre, esrc, b1, agg, stats);
  k_bn_relu<<<(NN * HH + 255) / 256, 256, 0, stream>>>(agg, stats, g1, be1,
                                                       nullptr, h1b);
  // ---- spatial layer 2 ----
  k_gemm<HH><<<NN / 16, 128, 0, stream>>>(h1b, W2t, hpre);
  k_aggregate<<<NN / 8, 256, 0, stream>>>(hpre, esrc, b2, agg, stats + 2 * HH);
  k_bn_relu<<<(NN * HH + 255) / 256, 256, 0, stream>>>(agg, stats + 2 * HH, g2, be2,
                                                       h2, nullptr);
  // ---- temporal branch (fused attention + GRU) ----
  k_temporal<<<NN / 16, 384, 0, stream>>>(x, ts, esrc, Wihb, Whhb, bih, bhh,
                                          beta, hT);
  // ---- head ----
  k_head<<<(NN + 255) / 256, 256, 0, stream>>>(h2, hT, Wo, bo, out);
}
